// WavAttention_81870666596608
// MI455X (gfx1250) — matrix operand order repaired and checked by reading, not compiled, not run on hardware
//
#include <hip/hip_runtime.h>
#include <hip/hip_bf16.h>

typedef _Float16 half_t;
typedef __attribute__((ext_vector_type(16))) _Float16 v16h;
typedef __attribute__((ext_vector_type(8)))  _Float16 v8h;
typedef __attribute__((ext_vector_type(8)))  float    v8f;

// Concatenate two 8-half vectors into a 16-half WMMA fragment.
__device__ inline v16h cat8(v8h lo, v8h hi) {
  return __builtin_shufflevector(lo, hi, 0, 1, 2, 3, 4, 5, 6, 7,
                                 8, 9, 10, 11, 12, 13, 14, 15);
}

// ---------------------------------------------------------------------------
// Fragment layouts used here (A/B: contiguous k, not the ISA table order), wave32, 16x16x32 f16:
//  A (16x32): lane l -> row m=l&15, h=l>>4; halves 0..15 = K 16h..16h+15
//             of row m -> contiguous 32B (two 16B chunks).
//  B (32x16): lane l -> col n=l&15, h=l>>4; halves 0..15 = K 16h..16h+15
//             of column n -> contiguous 32B when K is the fast axis.
//  C/D (16x16 f32): lane l, vgpr r -> M = r + 8*(l>>4), N = l&15.
// ---------------------------------------------------------------------------

// A fragment from row-major MxK memory (row stride lda halves, 16B-aligned)
__device__ inline v16h load_a_frag_g(const half_t* __restrict__ A, int lda) {
  int l = threadIdx.x & 31;
  const half_t* p = A + (size_t)(l & 15) * lda + 16 * (l >> 4);
  return cat8(*(const v8h*)p, *(const v8h*)(p + 8));
}

// ---------------------------------------------------------------------------
// f32 -> f16 convert
// ---------------------------------------------------------------------------
__global__ void cvt_f16_kernel(const float* __restrict__ src, half_t* __restrict__ dst, int n) {
  int i = blockIdx.x * blockDim.x + threadIdx.x;
  if (i < n) dst[i] = (half_t)src[i];
}

// ---------------------------------------------------------------------------
// Batched GEMM: C[b] = A(MxK) * B[b](KxN) + bias, f16 in / f32 acc.
// Block: 8 waves -> 64x128 tile; each wave computes a 32x32 quad (2x2 16x16
// tiles, 4 WMMAs per K-step, A/B fragments reused).  B is staged transposed
// in double-buffered LDS: one barrier per K-step, next tile's global loads
// issued before the current compute.
// Optional outputs: f32 (Cf), f16 same layout (Ch), f16 head-split transposed
// (CtH: [(b*8 + m/64)][n][m%64], one b128 store per lane).
// grid = (N/128, M/64, batch), block = (32,8)
// ---------------------------------------------------------------------------
__global__ void __launch_bounds__(256) gemm16_kernel(
    const half_t* __restrict__ A, const half_t* __restrict__ B,
    const float* __restrict__ bias, float* __restrict__ Cf,
    half_t* __restrict__ Ch, half_t* __restrict__ CtH,
    int M, int N, int K) {
  __shared__ half_t Bt[2][128][40];    // [buf][n][k], 80B row stride (16B aligned)
  int b  = blockIdx.z;
  int n0 = blockIdx.x * 128;
  int m0 = blockIdx.y * 64;
  int wave = threadIdx.y;              // 0..7
  int wm = wave >> 2;                  // 0..1 : 32-row half
  int wn = wave & 3;                   // 0..3 : 32-col quarter
  int tid = wave * 32 + threadIdx.x;
  int kld = tid >> 3;                  // 0..31 : K row to stage
  int ng  = tid & 7;                   // 0..7  : 16-wide N group
  int l = threadIdx.x & 31;
  const half_t* Bb = B + (size_t)b * K * N;

  v8f zero = {};
  v8f acc[2][2];
#pragma unroll
  for (int mi = 0; mi < 2; ++mi)
#pragma unroll
    for (int ni = 0; ni < 2; ++ni) acc[mi][ni] = zero;

  // prologue: stage K-step 0 into buffer 0
  {
    v8h c0 = *(const v8h*)(Bb + (size_t)kld * N + n0 + ng * 16);
    v8h c1 = *(const v8h*)(Bb + (size_t)kld * N + n0 + ng * 16 + 8);
#pragma unroll
    for (int u = 0; u < 8; ++u) {
      Bt[0][ng * 16 + u][kld]     = c0[u];
      Bt[0][ng * 16 + 8 + u][kld] = c1[u];
    }
  }
  for (int kk = 0, it = 0; kk < K; kk += 32, ++it) {
    int buf = it & 1;
    bool more = kk + 32 < K;
    v8h nv0, nv1;
    if (more) {
      nv0 = *(const v8h*)(Bb + (size_t)(kk + 32 + kld) * N + n0 + ng * 16);
      nv1 = *(const v8h*)(Bb + (size_t)(kk + 32 + kld) * N + n0 + ng * 16 + 8);
      if (kk + 64 < K)
        __builtin_prefetch(Bb + (size_t)(kk + 64 + kld) * N + n0 + ng * 16, 0, 3);
    }
    __syncthreads();   // prev reads of buf^1 done; stores to buf visible
    // fragments
    v16h am[2], bn[2];
#pragma unroll
    for (int mi = 0; mi < 2; ++mi)
      am[mi] = load_a_frag_g(A + (size_t)(m0 + wm * 32 + mi * 16) * K + kk, K);
#pragma unroll
    for (int ni = 0; ni < 2; ++ni) {
      const half_t* pb = &Bt[buf][(wn * 2 + ni) * 16 + (l & 15)][16 * (l >> 4)];
      bn[ni] = cat8(*(const v8h*)pb, *(const v8h*)(pb + 8));
    }
#pragma unroll
    for (int mi = 0; mi < 2; ++mi)
#pragma unroll
      for (int ni = 0; ni < 2; ++ni)
        acc[mi][ni] = __builtin_amdgcn_wmma_f32_16x16x32_f16(
            false, am[mi], false, bn[ni], (short)0, acc[mi][ni], false, false);
    if (more) {
#pragma unroll
      for (int u = 0; u < 8; ++u) {
        Bt[buf ^ 1][ng * 16 + u][kld]     = nv0[u];
        Bt[buf ^ 1][ng * 16 + 8 + u][kld] = nv1[u];
      }
    }
  }
  // epilogue
  int h = l >> 4, n = l & 15;
#pragma unroll
  for (int mi = 0; mi < 2; ++mi)
#pragma unroll
    for (int ni = 0; ni < 2; ++ni) {
      int mbase = m0 + wm * 32 + mi * 16;
      int nn = n0 + (wn * 2 + ni) * 16 + n;
      if (Cf || Ch) {
#pragma unroll
        for (int r = 0; r < 8; ++r) {
          int m = mbase + r + 8 * h;
          float v = acc[mi][ni][r] + bias[m];
          size_t idx = (size_t)b * M * N + (size_t)m * N + nn;
          if (Cf) Cf[idx] = v;
          if (Ch) Ch[idx] = (half_t)v;
        }
      }
      if (CtH) {    // head-split transposed: [(b*8 + m/64)][n][m%64]
        v8h pk;
#pragma unroll
        for (int r = 0; r < 8; ++r) {
          int m = mbase + r + 8 * h;
          pk[r] = (half_t)(acc[mi][ni][r] + bias[m]);
        }
        int mB = mbase + 8 * h;       // 8 consecutive m = 8 consecutive d
        size_t idx = (((size_t)b * 8 + (mB >> 6)) * N + nn) * 64 + (mB & 63);
        *(v8h*)(CtH + idx) = pk;
      }
    }
}

// ---------------------------------------------------------------------------
// conv_offset: depthwise 3x3 stride2 pad1 on q viewed (64,128,32,32) -> t(64,128,16,16)
// ---------------------------------------------------------------------------
__global__ void off_dw_kernel(const half_t* __restrict__ qh, const float* __restrict__ w,
                              const float* __restrict__ bias, float* __restrict__ t) {
  int bg = blockIdx.x, p = threadIdx.x;   // 64 blocks x 256 threads
  int b = bg >> 2, g = bg & 3;
  int oy = p >> 4, ox = p & 15;
  for (int cc = 0; cc < 128; ++cc) {
    const half_t* qb = qh + (size_t)(b * 512 + g * 128 + cc) * 1024;
    float acc = bias[cc];
#pragma unroll
    for (int dy = 0; dy < 3; ++dy) {
      int iy = 2 * oy + dy - 1;
      if (iy < 0 || iy > 31) continue;
#pragma unroll
      for (int dx = 0; dx < 3; ++dx) {
        int ix = 2 * ox + dx - 1;
        if (ix < 0 || ix > 31) continue;
        acc += (float)qb[iy * 32 + ix] * w[cc * 9 + dy * 3 + dx];
      }
    }
    t[((size_t)bg * 128 + cc) * 256 + p] = acc;
  }
}

// LN(channel) + exact GELU + 1x1 (128 -> 2) -> offset (64,2,256)
__global__ void off_lnpw_kernel(const float* __restrict__ t, const float* __restrict__ gam,
                                const float* __restrict__ bet, const float* __restrict__ pw,
                                float* __restrict__ offb) {
  int bg = blockIdx.x, p = threadIdx.x;
  const float* tb = t + (size_t)bg * 128 * 256;
  float s = 0.f, s2 = 0.f;
  for (int cc = 0; cc < 128; ++cc) { float v = tb[cc * 256 + p]; s += v; s2 += v * v; }
  float mu = s * (1.f / 128.f);
  float var = s2 * (1.f / 128.f) - mu * mu;
  float rstd = rsqrtf(var + 1e-5f);
  float a0 = 0.f, a1 = 0.f;
  for (int cc = 0; cc < 128; ++cc) {
    float u  = (tb[cc * 256 + p] - mu) * rstd * gam[cc] + bet[cc];
    float gl = 0.5f * u * (1.f + erff(u * 0.70710678118f));
    a0 += gl * pw[cc];
    a1 += gl * pw[128 + cc];
  }
  offb[(size_t)bg * 512 + p]       = a0;
  offb[(size_t)bg * 512 + 256 + p] = a1;
}

// ---------------------------------------------------------------------------
// ab path: 1D depthwise (16ch,k=3,s=2,p=1) + LN + GELU + 1x1 -> ab_offset (64,2,16)
// ---------------------------------------------------------------------------
__global__ void ab_kernel(const float* __restrict__ sz, const float* __restrict__ w,
                          const float* __restrict__ bias, const float* __restrict__ gam,
                          const float* __restrict__ bet, const float* __restrict__ pw,
                          float* __restrict__ abo) {
  int bg = blockIdx.x, ol = threadIdx.x;    // 64 blocks x 16 threads
  const float* zb = sz + (size_t)bg * 512;  // (16,32)
  float sv[16];
  float s = 0.f, s2 = 0.f;
  for (int cc = 0; cc < 16; ++cc) {
    float acc = bias[cc];
#pragma unroll
    for (int k = 0; k < 3; ++k) {
      int il = 2 * ol + k - 1;
      if (il >= 0 && il < 32) acc += zb[cc * 32 + il] * w[cc * 3 + k];
    }
    sv[cc] = acc; s += acc; s2 += acc * acc;
  }
  float mu = s * 0.0625f;
  float var = s2 * 0.0625f - mu * mu;
  float rstd = rsqrtf(var + 1e-5f);
  float a0 = 0.f, a1 = 0.f;
  for (int cc = 0; cc < 16; ++cc) {
    float u  = (sv[cc] - mu) * rstd * gam[cc] + bet[cc];
    float gl = 0.5f * u * (1.f + erff(u * 0.70710678118f));
    a0 += gl * pw[cc];
    a1 += gl * pw[16 + cc];
  }
  abo[(size_t)bg * 32 + ol]      = a0;
  abo[(size_t)bg * 32 + 16 + ol] = a1;
}

// ---------------------------------------------------------------------------
// 16x16 inverse DFT (real part) -> gradient -> direction -> sampling positions
// ---------------------------------------------------------------------------
__global__ void fftpos_kernel(const float* __restrict__ offb, const float* __restrict__ abo,
                              float* __restrict__ pos) {
  __shared__ float dm[256];
  int bg = blockIdx.x, tid = threadIdx.x;
  int i = tid >> 4, j = tid & 15;
  const float* o0 = offb + (size_t)bg * 512;
  const float* o1 = o0 + 256;
  float acc = 0.f;
  const float w0 = 6.283185307179586f / 16.f;
  for (int k = 0; k < 16; ++k)
    for (int l2 = 0; l2 < 16; ++l2) {
      float th = w0 * (float)((i * k + j * l2) & 15);
      float sn, cs; __sincosf(th, &sn, &cs);
      acc += o0[k * 16 + l2] * cs - o1[k * 16 + l2] * sn;
    }
  dm[tid] = acc * (1.f / 256.f);
  __syncthreads();
  float gy = (i == 0) ? dm[16 + j] - dm[j]
           : (i == 15) ? dm[15 * 16 + j] - dm[14 * 16 + j]
           : 0.5f * (dm[(i + 1) * 16 + j] - dm[(i - 1) * 16 + j]);
  float gx = (j == 0) ? dm[i * 16 + 1] - dm[i * 16]
           : (j == 15) ? dm[i * 16 + 15] - dm[i * 16 + 14]
           : 0.5f * (dm[i * 16 + j + 1] - dm[i * 16 + j - 1]);
  float mag = sqrtf(gx * gx + gy * gy);
  float inv = 0.1f / (mag + 1e-6f);
  float ry = ((float)i + 0.5f) * (2.f / 15.f) - 1.f;
  float rx = ((float)j + 0.5f) * (2.f / 15.f) - 1.f;
  float a0 = abo[(size_t)bg * 32 + j];
  float a1 = abo[(size_t)bg * 32 + 16 + j];
  pos[((size_t)bg * 256 + tid) * 2 + 0] = fminf(fmaxf(-gx * inv + a0 + ry, -1.f), 1.f);
  pos[((size_t)bg * 256 + tid) * 2 + 1] = fminf(fmaxf(-gy * inv + a1 + rx, -1.f), 1.f);
}

// ---------------------------------------------------------------------------
// bilinear grid_sample (align_corners=True, zero pad) -> xs f16
// ---------------------------------------------------------------------------
__global__ void sample_kernel(const float* __restrict__ x, const float* __restrict__ pos,
                              half_t* __restrict__ xsh) {
  int bg = blockIdx.x, p = threadIdx.x;
  int b = bg >> 2, g = bg & 3;
  float p0 = pos[((size_t)bg * 256 + p) * 2 + 0];
  float p1 = pos[((size_t)bg * 256 + p) * 2 + 1];
  float gxs = (p1 + 1.f) * 0.5f * 31.f;
  float gys = (p0 + 1.f) * 0.5f * 31.f;
  float x0 = floorf(gxs), y0 = floorf(gys);
  float wt[4]; int xi[4], yi[4];
  int q = 0;
#pragma unroll
  for (int dy = 0; dy < 2; ++dy)
#pragma unroll
    for (int dx = 0; dx < 2; ++dx) {
      float xf = x0 + dx, yf = y0 + dy;
      float w = (1.f - fabsf(gxs - xf)) * (1.f - fabsf(gys - yf));
      bool valid = (xf >= 0.f) && (xf < 32.f) && (yf >= 0.f) && (yf < 32.f);
      wt[q] = valid ? w : 0.f;
      xi[q] = (int)fminf(fmaxf(xf, 0.f), 31.f);
      yi[q] = (int)fminf(fmaxf(yf, 0.f), 31.f);
      ++q;
    }
  for (int cc = 0; cc < 128; ++cc) {
    const float* xb = x + (size_t)(b * 512 + g * 128 + cc) * 1024;
    float acc = 0.f;
#pragma unroll
    for (int t = 0; t < 4; ++t) acc += xb[yi[t] * 32 + xi[t]] * wt[t];
    xsh[((size_t)(b * 512) + g * 128 + cc) * 256 + p] = (half_t)acc;
  }
}

// ---------------------------------------------------------------------------
// lepe: depthwise 3x3 stride1 pad1 on q (16,512,32,32) -> f16
// ---------------------------------------------------------------------------
__global__ void lepe_kernel(const half_t* __restrict__ qh, const float* __restrict__ w,
                            const float* __restrict__ bias, half_t* __restrict__ lep) {
  size_t idx = (size_t)blockIdx.x * blockDim.x + threadIdx.x;
  if (idx >= (size_t)16 * 512 * 1024) return;
  int hw = (int)(idx & 1023);
  int c  = (int)((idx >> 10) & 511);
  int y = hw >> 5, xx = hw & 31;
  const half_t* qb = qh + (idx - hw);
  float acc = bias[c];
#pragma unroll
  for (int dy = 0; dy < 3; ++dy) {
    int iy = y + dy - 1;
    if (iy < 0 || iy > 31) continue;
#pragma unroll
    for (int dx = 0; dx < 3; ++dx) {
      int ix = xx + dx - 1;
      if (ix < 0 || ix > 31) continue;
      acc += (float)qb[iy * 32 + ix] * w[c * 9 + dy * 3 + dx];
    }
  }
  lep[idx] = (half_t)acc;
}

// ---------------------------------------------------------------------------
// Fused attention: block = 4 waves, each wave one (bh, 16-row m-tile).
// q^T and k^T come pre-transposed from the producing GEMMs ([bh][m][d] /
// [bh][n][d]) so every fragment load is a pair of 128-bit global loads.
//   S = q^T k (32 WMMAs/wave) -> softmax in registers (shfl_xor over 16-lane
//   halves) -> P staged f16 in LDS -> out = P v^T (32 WMMAs/wave) -> +lepe.
// grid = (1024/64, 128), block = (32,4)
// ---------------------------------------------------------------------------
__global__ void __launch_bounds__(128) attn_kernel(
    const half_t* __restrict__ qt, const half_t* __restrict__ kt,
    const half_t* __restrict__ vh, const half_t* __restrict__ lep,
    half_t* __restrict__ outh) {
  __shared__ half_t P[4][16][272];    // per-wave probabilities, 544B row stride
  int bh = blockIdx.y;                // 0..127
  int b = bh >> 3, head = bh & 7;
  int wave = threadIdx.y;             // 0..3
  int l = threadIdx.x;
  int m0 = blockIdx.x * 64 + wave * 16;
  const half_t* qrow  = qt + ((size_t)bh * 1024 + m0) * 64;         // [m][d]
  const half_t* krow  = kt + (size_t)bh * 256 * 64;                 // [n][d]
  const half_t* vbase = vh + ((size_t)b * 512 + head * 64) * 256;   // [d][n]

  v8f zero = {};
  v8f sacc[16];
#pragma unroll
  for (int i = 0; i < 16; ++i) sacc[i] = zero;

  // A fragments of q^T (row-major m x d): two b128 each
  v16h afr[2];
#pragma unroll
  for (int kc = 0; kc < 2; ++kc) afr[kc] = load_a_frag_g(qrow + kc * 32, 64);

  // S = q^T k : B fragments straight from k^T (column n = row of k^T)
#pragma unroll
  for (int nt = 0; nt < 16; ++nt)
#pragma unroll
    for (int kc = 0; kc < 2; ++kc) {
      const half_t* p = krow + (size_t)(nt * 16 + (l & 15)) * 64 + kc * 32 + 16 * (l >> 4);
      v16h bfr = cat8(*(const v8h*)p, *(const v8h*)(p + 8));
      sacc[nt] = __builtin_amdgcn_wmma_f32_16x16x32_f16(false, afr[kc], false, bfr,
                                                        (short)0, sacc[nt], false, false);
    }

  // softmax over n (scale, row-max, exp, row-sum; rows span 16-lane halves)
  const float scale = 0.125f;  // D^-0.5
  float rnorm[8], rmax[8];
#pragma unroll
  for (int r = 0; r < 8; ++r) {
    float mx = -3.4e38f;
#pragma unroll
    for (int nt = 0; nt < 16; ++nt) mx = fmaxf(mx, sacc[nt][r]);
    for (int s = 8; s >= 1; s >>= 1) mx = fmaxf(mx, __shfl_xor(mx, s, 32));
    rmax[r] = mx * scale;
  }
#pragma unroll
  for (int r = 0; r < 8; ++r) {
    float sm = 0.f;
#pragma unroll
    for (int nt = 0; nt < 16; ++nt) {
      float p = __expf(sacc[nt][r] * scale - rmax[r]);
      sacc[nt][r] = p;
      sm += p;
    }
    for (int s = 8; s >= 1; s >>= 1) sm += __shfl_xor(sm, s, 32);
    rnorm[r] = 1.f / sm;
  }
  // stage P (f16) in this wave's LDS region
  {
    int h = l >> 4, nn = l & 15;
#pragma unroll
    for (int nt = 0; nt < 16; ++nt)
#pragma unroll
      for (int r = 0; r < 8; ++r)
        P[wave][r + 8 * h][nt * 16 + nn] = (half_t)(sacc[nt][r] * rnorm[r]);
  }
  __syncthreads();

  // out = P (16x256) * v^T (256x64): A frags from LDS (own region), B frags
  // straight from global (v is [d][n]: 16 contiguous halves -> 2x b128).
  v8f oacc[4];
#pragma unroll
  for (int dt = 0; dt < 4; ++dt) oacc[dt] = zero;
#pragma unroll
  for (int kb = 0; kb < 8; ++kb) {
    v16h a;
    {
      const half_t* p = &P[wave][l & 15][kb * 32 + 16 * (l >> 4)];
      a = cat8(*(const v8h*)p, *(const v8h*)(p + 8));
    }
#pragma unroll
    for (int dt = 0; dt < 4; ++dt) {
      const half_t* p = vbase + (size_t)(dt * 16 + (l & 15)) * 256 + kb * 32 + 16 * (l >> 4);
      v16h bfr = cat8(*(const v8h*)p, *(const v8h*)(p + 8));
      oacc[dt] = __builtin_amdgcn_wmma_f32_16x16x32_f16(false, a, false, bfr,
                                                        (short)0, oacc[dt], false, false);
    }
  }
  // epilogue: rows = m, cols = d_local; add lepe, store f16
  {
    int h = l >> 4, d = l & 15;
#pragma unroll
    for (int dt = 0; dt < 4; ++dt)
#pragma unroll
      for (int r = 0; r < 8; ++r) {
        int m = m0 + r + 8 * h;
        int c = head * 64 + dt * 16 + d;
        size_t idx = ((size_t)b * 512 + c) * 1024 + m;
        outh[idx] = (half_t)(oacc[dt][r] + (float)lep[idx]);
      }
  }
}

// ---------------------------------------------------------------------------
extern "C" void kernel_launch(void* const* d_in, const int* in_sizes, int n_in,
                              void* d_out, int out_size, void* d_ws, size_t ws_size,
                              hipStream_t stream) {
  const float* x        = (const float*)d_in[0];
  const float* strz     = (const float*)d_in[1];
  const float* off_dw_w = (const float*)d_in[2];
  const float* off_dw_b = (const float*)d_in[3];
  const float* off_ln_g = (const float*)d_in[4];
  const float* off_ln_b = (const float*)d_in[5];
  const float* off_pw_w = (const float*)d_in[6];
  const float* ab_dw_w  = (const float*)d_in[7];
  const float* ab_dw_b  = (const float*)d_in[8];
  const float* ab_ln_g  = (const float*)d_in[9];
  const float* ab_ln_b  = (const float*)d_in[10];
  const float* ab_pw_w  = (const float*)d_in[11];
  const float* wq = (const float*)d_in[12]; const float* bq = (const float*)d_in[13];
  const float* wk = (const float*)d_in[14]; const float* bk = (const float*)d_in[15];
  const float* wv = (const float*)d_in[16]; const float* bv = (const float*)d_in[17];
  const float* wo = (const float*)d_in[18]; const float* bo = (const float*)d_in[19];
  const float* rpe_w = (const float*)d_in[20]; const float* rpe_b = (const float*)d_in[21];
  float* y = (float*)d_out;

  char* p = (char*)d_ws;
  auto alloc = [&](size_t bytes) -> void* {
    void* r = (void*)p;
    p += (bytes + 255) & ~(size_t)255;
    return r;
  };
  const size_t QN = (size_t)16 * 512 * 1024;   // q / out elements
  const size_t SN = (size_t)16 * 512 * 256;    // xs / k / v elements
  half_t* xh   = (half_t*)alloc(QN * 2);       // reused as out_h after proj_q
  half_t* wqh  = (half_t*)alloc(512 * 512 * 2);
  half_t* wkh  = (half_t*)alloc(512 * 512 * 2);
  half_t* wvh  = (half_t*)alloc(512 * 512 * 2);
  half_t* woh  = (half_t*)alloc(512 * 512 * 2);
  half_t* qh   = (half_t*)alloc(QN * 2);       // q, channel-major (dwconvs)
  half_t* qtb  = (half_t*)alloc(QN * 2);       // q^T, [bh][m][d] (attention A)
  float*  tbuf = (float*)alloc((size_t)64 * 128 * 256 * 4);
  float*  offb = (float*)alloc((size_t)64 * 2 * 256 * 4);
  float*  abo  = (float*)alloc((size_t)64 * 2 * 16 * 4);
  float*  posb = (float*)alloc((size_t)64 * 256 * 2 * 4);
  half_t* xsh  = (half_t*)alloc(SN * 2);
  half_t* ktb  = (half_t*)alloc(SN * 2);       // k^T, [bh][n][d] (attention B)
  half_t* vhb  = (half_t*)alloc(SN * 2);       // v, channel-major
  half_t* lepb = (half_t*)alloc(QN * 2);
  half_t* outh = xh;                           // alias: xh dead after proj_q

  // 1) convert inputs to f16
  cvt_f16_kernel<<<(int)((QN + 255) / 256), 256, 0, stream>>>(x, xh, (int)QN);
  cvt_f16_kernel<<<1024, 256, 0, stream>>>(wq, wqh, 512 * 512);
  cvt_f16_kernel<<<1024, 256, 0, stream>>>(wk, wkh, 512 * 512);
  cvt_f16_kernel<<<1024, 256, 0, stream>>>(wv, wvh, 512 * 512);
  cvt_f16_kernel<<<1024, 256, 0, stream>>>(wo, woh, 512 * 512);

  dim3 gblk(32, 8);
  // 2) q = wq * x + bq  (M=512,N=1024,K=512): channel-major f16 + head-transposed f16
  gemm16_kernel<<<dim3(8, 8, 16), gblk, 0, stream>>>(wqh, xh, bq, nullptr, qh, qtb, 512, 1024, 512);
  // 3) offset path
  off_dw_kernel<<<64, 256, 0, stream>>>(qh, off_dw_w, off_dw_b, tbuf);
  off_lnpw_kernel<<<64, 256, 0, stream>>>(tbuf, off_ln_g, off_ln_b, off_pw_w, offb);
  ab_kernel<<<64, 16, 0, stream>>>(strz, ab_dw_w, ab_dw_b, ab_ln_g, ab_ln_b, ab_pw_w, abo);
  fftpos_kernel<<<64, 256, 0, stream>>>(offb, abo, posb);
  sample_kernel<<<64, 256, 0, stream>>>(x, posb, xsh);
  // 4) k (head-transposed only), v (channel-major)  (M=512,N=256,K=512)
  gemm16_kernel<<<dim3(2, 8, 16), gblk, 0, stream>>>(wkh, xsh, bk, nullptr, nullptr, ktb, 512, 256, 512);
  gemm16_kernel<<<dim3(2, 8, 16), gblk, 0, stream>>>(wvh, xsh, bv, nullptr, vhb, nullptr, 512, 256, 512);
  // 5) lepe
  lepe_kernel<<<(int)((QN + 255) / 256), 256, 0, stream>>>(qh, rpe_w, rpe_b, lepb);
  // 6) fused attention + lepe -> out (f16)
  attn_kernel<<<dim3(16, 128), dim3(32, 4), 0, stream>>>(qtb, ktb, vhb, lepb, outh);
  // 7) y = wo * out + bo  -> d_out (f32)
  gemm16_kernel<<<dim3(8, 8, 16), gblk, 0, stream>>>(woh, outh, bo, y, nullptr, nullptr, 512, 1024, 512);
}
